// Attention_17884243821438
// MI455X (gfx1250) — compile-verified, matrix operand order checked by reading, not run on hardware
//
#include <hip/hip_runtime.h>

// ---------- types ----------
typedef __attribute__((ext_vector_type(16))) _Float16 v16h;
typedef __attribute__((ext_vector_type(8)))  _Float16 v8h;
typedef __attribute__((ext_vector_type(8)))  float    v8f;

#define B_   4
#define N_   2048
#define D_   512
#define H_   8
#define KVH_ 2
#define DH_  64
#define BN_  (B_ * N_)   // 8192 tokens

// ---------- WMMA helpers ----------
__device__ __forceinline__ v8f wmma_f16(v16h a, v16h b, v8f c) {
  // D = A(16x32 f16) x B(32x16 f16) + C(16x16 f32)
  return __builtin_amdgcn_wmma_f32_16x16x32_f16(
      /*neg_a=*/false, a, /*neg_b=*/false, b,
      /*c_mod=*/(short)0, c, /*reuse_a=*/false, /*reuse_b=*/false);
}

// Load a 16x32 f16 fragment (A-layout; identical striping assumed for B^T rows)
// from a row-major matrix with leading dimension `ld` (in elements).
// Lane (r = lane&15, g = lane>>4) holds row r, K = {g*8..g*8+7, 16+g*8..16+g*8+7}.
__device__ __forceinline__ v16h load_frag(const _Float16* __restrict__ base,
                                          int ld, int lane) {
  const int r = lane & 15, g = lane >> 4;
  const _Float16* p = base + (size_t)r * ld + g * 8;
  v8h lo = *(const v8h*)(p);
  v8h hi = *(const v8h*)(p + 16);
  v16h f;
#pragma unroll
  for (int i = 0; i < 8; ++i) { f[i] = lo[i]; f[i + 8] = hi[i]; }
  return f;
}

// ---------- 1) LayerNorm: x (b,d,n) -> xn (b*n, d) f32 + f16 ----------
__global__ __launch_bounds__(256) void layernorm_k(
    const float* __restrict__ x, const float* __restrict__ gamma,
    const float* __restrict__ beta, float* __restrict__ xnf,
    _Float16* __restrict__ xnh) {
  const int t = blockIdx.x * 256 + threadIdx.x;  // token id, 8192 total
  const int b = t >> 11, pos = t & (N_ - 1);
  const float* xb = x + (size_t)b * D_ * N_ + pos;  // stride N_ over d, coalesced over pos
  float s = 0.f, ss = 0.f;
  for (int d = 0; d < D_; ++d) {
    float v = xb[(size_t)d * N_];
    s += v; ss += v * v;
  }
  const float mu  = s * (1.f / D_);
  const float var = ss * (1.f / D_) - mu * mu;
  const float rst = rsqrtf(var + 1e-5f);
  float*     o  = xnf + (size_t)t * D_;
  _Float16*  oh = xnh + (size_t)t * D_;
  for (int d = 0; d < D_; ++d) {
    float v = (xb[(size_t)d * N_] - mu) * rst * gamma[d] + beta[d];
    o[d]  = v;
    oh[d] = (_Float16)v;
  }
}

// ---------- 2) Weight convert + transpose: W (K x N) f32 -> W^T (N x K) f16 ----------
__global__ __launch_bounds__(256) void transpose_w_k(
    const float* __restrict__ W, _Float16* __restrict__ WT, int K, int N) {
  const int idx = blockIdx.x * 256 + threadIdx.x;
  if (idx >= K * N) return;
  const int k = idx / N, n = idx % N;
  WT[(size_t)n * K + k] = (_Float16)W[(size_t)k * N + n];
}

// ---------- 3) Generic WMMA GEMM: C = A(MxK) * B(KxN), B given transposed ----------
// EPI 0: store f16 row-major MxN.
// EPI 1: out-proj epilogue: +resid(row,col) +bias(col), store f32 transposed (b,d,n).
template <int EPI>
__global__ __launch_bounds__(256) void gemm_wmma_k(
    const _Float16* __restrict__ A, const _Float16* __restrict__ BT,
    int M, int N, int K, _Float16* __restrict__ Ch,
    const float* __restrict__ resid, const float* __restrict__ bias,
    float* __restrict__ outT) {
  const int lane = threadIdx.x & 31;
  const int w    = threadIdx.x >> 5;             // 8 waves
  const int m0   = blockIdx.x * 64  + (w & 1) * 32;
  const int n0   = blockIdx.y * 128 + (w >> 1) * 32;

  const v8f z = {0.f, 0.f, 0.f, 0.f, 0.f, 0.f, 0.f, 0.f};
  v8f acc[2][2];
  acc[0][0] = z; acc[0][1] = z; acc[1][0] = z; acc[1][1] = z;

  for (int k0 = 0; k0 < K; k0 += 32) {
    v16h a0 = load_frag(A  + (size_t)(m0     ) * K + k0, K, lane);
    v16h a1 = load_frag(A  + (size_t)(m0 + 16) * K + k0, K, lane);
    v16h b0 = load_frag(BT + (size_t)(n0     ) * K + k0, K, lane);
    v16h b1 = load_frag(BT + (size_t)(n0 + 16) * K + k0, K, lane);
    acc[0][0] = wmma_f16(a0, b0, acc[0][0]);
    acc[0][1] = wmma_f16(a0, b1, acc[0][1]);
    acc[1][0] = wmma_f16(a1, b0, acc[1][0]);
    acc[1][1] = wmma_f16(a1, b1, acc[1][1]);
  }

  const int c = lane & 15, g = lane >> 4;
#pragma unroll
  for (int ti = 0; ti < 2; ++ti)
#pragma unroll
    for (int tj = 0; tj < 2; ++tj)
#pragma unroll
      for (int p = 0; p < 8; ++p) {
        const int row = m0 + ti * 16 + g * 8 + p;
        const int col = n0 + tj * 16 + c;
        float v = acc[ti][tj][p];
        if (EPI == 0) {
          Ch[(size_t)row * N + col] = (_Float16)v;
        } else {
          v += resid[(size_t)row * N + col] + bias[col];
          const int bb = row >> 11, pos = row & (N_ - 1);
          outT[((size_t)(bb * D_ + col)) * N_ + pos] = v;
        }
      }
}

// ---------- 4) xPos rotary ----------
__device__ __forceinline__ void xpos_coeffs(int pos, int j, float& cs, float& sn,
                                            float& scl) {
  const float fpos = (float)pos;
  const float invf = __powf(10000.f, -(float)(2 * j) * (1.f / (float)DH_));
  const float fr   = fpos * invf;
  cs = __cosf(fr);
  sn = __sinf(fr);
  const float bs = ((float)(2 * j) + 0.4f * DH_) * (1.f / (1.4f * DH_));
  const float pw = (fpos - (float)(N_ / 2)) * (1.f / 4096.f);
  scl = __powf(bs, pw);
}

__global__ __launch_bounds__(256) void rotary_q_k(
    const _Float16* __restrict__ qraw, _Float16* __restrict__ qh) {
  const int idx = blockIdx.x * 256 + threadIdx.x;  // b*h*n*dh = 4M
  const int i   = idx & 63;
  const int pos = (idx >> 6) & (N_ - 1);
  const int h   = (idx >> 17) & (H_ - 1);
  const int b   = idx >> 20;
  const size_t tok = (size_t)(b * N_ + pos);
  const _Float16* qp = qraw + tok * (H_ * DH_) + h * DH_;
  const float qv = (float)qp[i];
  const float pr = (i < 32) ? -(float)qp[i + 32] : (float)qp[i - 32];
  float cs, sn, scl;
  xpos_coeffs(pos, i & 31, cs, sn, scl);
  const float out = (qv * cs + pr * sn) * scl * 0.125f;  // fold 1/sqrt(dh)
  qh[(((size_t)(b * H_ + h) * N_) + pos) * DH_ + i] = (_Float16)out;
}

__global__ __launch_bounds__(256) void rotary_kv_k(
    const _Float16* __restrict__ kvraw, _Float16* __restrict__ kh,
    _Float16* __restrict__ vT) {
  const int idx = blockIdx.x * 256 + threadIdx.x;  // b*kvh*n*dh = 1M
  const int i   = idx & 63;
  const int pos = (idx >> 6) & (N_ - 1);
  const int kv  = (idx >> 17) & 1;
  const int b   = idx >> 18;
  const size_t tok = (size_t)(b * N_ + pos);
  const _Float16* kp = kvraw + tok * 256 + kv * DH_;
  const _Float16* vp = kvraw + tok * 256 + 128 + kv * DH_;
  const float kvv = (float)kp[i];
  const float pr  = (i < 32) ? -(float)kp[i + 32] : (float)kp[i - 32];
  float cs, sn, scl;
  xpos_coeffs(pos, i & 31, cs, sn, scl);
  const float out = (kvv * cs + pr * sn) / scl;
  kh[(((size_t)(b * KVH_ + kv) * N_) + pos) * DH_ + i] = (_Float16)out;
  vT[(((size_t)(b * KVH_ + kv) * DH_) + i) * N_ + pos] = vp[i];
}

// ---------- 5) Flash attention: per (b,h), wave = 16 query rows ----------
__global__ __launch_bounds__(256) void flash_k(
    const _Float16* __restrict__ Q,   // (b, h, n, dh)
    const _Float16* __restrict__ Kt,  // (b, kvh, n, dh)
    const _Float16* __restrict__ VT,  // (b, kvh, dh, n)
    _Float16* __restrict__ O) {       // (b, n, h*dh)
  __shared__ __align__(16) _Float16 lds[8 * 16 * 32];  // per-wave 16x32 probs tile
  const int lane = threadIdx.x & 31;
  const int w    = threadIdx.x >> 5;
  const int h    = blockIdx.y, b = blockIdx.z;
  const int kv   = h & (KVH_ - 1);                     // tile(k, r): head h -> h % kvh
  const int qRow0 = blockIdx.x * 128 + w * 16;

  const _Float16* qptr  = Q  + (((size_t)(b * H_ + h) * N_) + qRow0) * DH_;
  const _Float16* kbase = Kt + ((size_t)(b * KVH_ + kv) * N_) * DH_;
  const _Float16* vbase = VT + ((size_t)(b * KVH_ + kv) * DH_) * N_;

  const v16h aq0 = load_frag(qptr,      DH_, lane);    // K dims  0..31
  const v16h aq1 = load_frag(qptr + 32, DH_, lane);    // K dims 32..63

  const v8f z = {0.f, 0.f, 0.f, 0.f, 0.f, 0.f, 0.f, 0.f};
  v8f oacc[4]; oacc[0] = z; oacc[1] = z; oacc[2] = z; oacc[3] = z;
  float mrow[8], lrow[8];
#pragma unroll
  for (int p = 0; p < 8; ++p) { mrow[p] = -1e30f; lrow[p] = 0.f; }

  const int c = lane & 15, g = lane >> 4;
  _Float16* wl = lds + w * 512;

  for (int kb = 0; kb < N_; kb += 32) {
    // ---- scores: two 16x16 f32 tiles covering 32 keys ----
    v8f s0 = z, s1 = z;
    {
      const _Float16* kp0 = kbase + (size_t)kb * DH_;
      s0 = wmma_f16(aq0, load_frag(kp0,      DH_, lane), s0);
      s0 = wmma_f16(aq1, load_frag(kp0 + 32, DH_, lane), s0);
      const _Float16* kp1 = kbase + (size_t)(kb + 16) * DH_;
      s1 = wmma_f16(aq0, load_frag(kp1,      DH_, lane), s1);
      s1 = wmma_f16(aq1, load_frag(kp1 + 32, DH_, lane), s1);
    }
    // ---- online softmax: row reductions across the 16-lane C-layout halves ----
#pragma unroll
    for (int p = 0; p < 8; ++p) {
      float rmax = fmaxf(s0[p], s1[p]);
      rmax = fmaxf(rmax, __shfl_xor(rmax, 1, 32));
      rmax = fmaxf(rmax, __shfl_xor(rmax, 2, 32));
      rmax = fmaxf(rmax, __shfl_xor(rmax, 4, 32));
      rmax = fmaxf(rmax, __shfl_xor(rmax, 8, 32));
      const float nm = fmaxf(mrow[p], rmax);
      const float sc = __expf(mrow[p] - nm);
      const float p0 = __expf(s0[p] - nm);
      const float p1 = __expf(s1[p] - nm);
      wl[(g * 8 + p) * 32 + c]      = (_Float16)p0;   // row-major 16x32 probs tile
      wl[(g * 8 + p) * 32 + 16 + c] = (_Float16)p1;
      float rs = p0 + p1;
      rs += __shfl_xor(rs, 1, 32);
      rs += __shfl_xor(rs, 2, 32);
      rs += __shfl_xor(rs, 4, 32);
      rs += __shfl_xor(rs, 8, 32);
      lrow[p] = lrow[p] * sc + rs;
      mrow[p] = nm;
#pragma unroll
      for (int j = 0; j < 4; ++j) oacc[j][p] *= sc;
    }
    // ---- reload probs as A-fragment, accumulate AV over dh in 4 tiles of 16 ----
    const v16h pa = load_frag(wl, 32, lane);
#pragma unroll
    for (int j = 0; j < 4; ++j) {
      const v16h bv = load_frag(vbase + (size_t)(j * 16) * N_ + kb, N_, lane);
      oacc[j] = wmma_f16(pa, bv, oacc[j]);
    }
  }

#pragma unroll
  for (int p = 0; p < 8; ++p) {
    const float inv = 1.f / lrow[p];
    const size_t row = (size_t)(b * N_ + qRow0 + g * 8 + p);
#pragma unroll
    for (int j = 0; j < 4; ++j)
      O[row * (H_ * DH_) + h * DH_ + j * 16 + c] = (_Float16)(oacc[j][p] * inv);
  }
}

// ---------- launch ----------
extern "C" void kernel_launch(void* const* d_in, const int* in_sizes, int n_in,
                              void* d_out, int out_size, void* d_ws, size_t ws_size,
                              hipStream_t stream) {
  const float* x     = (const float*)d_in[0];
  const float* gamma = (const float*)d_in[1];
  const float* beta  = (const float*)d_in[2];
  const float* Wq    = (const float*)d_in[3];
  const float* Wkv   = (const float*)d_in[4];
  const float* Wout  = (const float*)d_in[5];
  const float* bout  = (const float*)d_in[6];
  float* y = (float*)d_out;

  char* wsp = (char*)d_ws;
  auto take = [&](size_t bytes) {
    void* p = (void*)wsp;
    wsp += (bytes + 255) & ~(size_t)255;
    return p;
  };
  float*    xnf   = (float*)take((size_t)BN_ * D_ * 4);
  _Float16* xnh   = (_Float16*)take((size_t)BN_ * D_ * 2);
  _Float16* WqT   = (_Float16*)take((size_t)D_ * D_ * 2);
  _Float16* WkvT  = (_Float16*)take((size_t)256 * D_ * 2);
  _Float16* WoutT = (_Float16*)take((size_t)D_ * D_ * 2);
  _Float16* qraw  = (_Float16*)take((size_t)BN_ * D_ * 2);
  _Float16* kvraw = (_Float16*)take((size_t)BN_ * 256 * 2);
  _Float16* qh    = (_Float16*)take((size_t)B_ * H_ * N_ * DH_ * 2);
  _Float16* kh    = (_Float16*)take((size_t)B_ * KVH_ * N_ * DH_ * 2);
  _Float16* vT    = (_Float16*)take((size_t)B_ * KVH_ * N_ * DH_ * 2);
  _Float16* ao    = (_Float16*)take((size_t)BN_ * D_ * 2);

  layernorm_k<<<BN_ / 256, 256, 0, stream>>>(x, gamma, beta, xnf, xnh);
  transpose_w_k<<<(D_ * D_ + 255) / 256, 256, 0, stream>>>(Wq, WqT, D_, D_);
  transpose_w_k<<<(D_ * 256 + 255) / 256, 256, 0, stream>>>(Wkv, WkvT, D_, 256);
  transpose_w_k<<<(D_ * D_ + 255) / 256, 256, 0, stream>>>(Wout, WoutT, D_, D_);

  gemm_wmma_k<0><<<dim3(BN_ / 64, D_ / 128), 256, 0, stream>>>(
      xnh, WqT, BN_, D_, D_, qraw, nullptr, nullptr, nullptr);
  gemm_wmma_k<0><<<dim3(BN_ / 64, 256 / 128), 256, 0, stream>>>(
      xnh, WkvT, BN_, 256, D_, kvraw, nullptr, nullptr, nullptr);

  rotary_q_k<<<(B_ * H_ * N_ * DH_) / 256, 256, 0, stream>>>(qraw, qh);
  rotary_kv_k<<<(B_ * KVH_ * N_ * DH_) / 256, 256, 0, stream>>>(kvraw, kh, vT);

  flash_k<<<dim3(N_ / 128, H_, B_), 256, 0, stream>>>(qh, kh, vT, ao);

  gemm_wmma_k<1><<<dim3(BN_ / 64, D_ / 128), 256, 0, stream>>>(
      ao, WoutT, BN_, D_, D_, nullptr, xnf, bout, y);
}
